// GraphReasoningBlock_88046829568734
// MI455X (gfx1250) — compile-verified, matrix operand order checked by reading, not run on hardware
//
#include <hip/hip_runtime.h>

typedef _Float16 h16;
typedef __attribute__((ext_vector_type(8)))  _Float16 v8h;
typedef __attribute__((ext_vector_type(16))) _Float16 v16h;
typedef __attribute__((ext_vector_type(8)))  float    v8f;

#define WMMA_F32_F16(a, b, c) \
  __builtin_amdgcn_wmma_f32_16x16x32_f16(false, (a), false, (b), (short)0, (c), false, false)

#if defined(__gfx1250__) && __has_builtin(__builtin_amdgcn_tensor_load_to_lds)
#define USE_TDM 1
#else
#define USE_TDM 0
#endif

static constexpr int kB = 16;     // batch
static constexpr int kN = 1024;   // nodes
static constexpr int kC = 768;    // channels
static constexpr int kD = 384;    // head dim
static constexpr int kMN = kB * kN;  // 16384 flattened rows

// Build a 16-bit A/B WMMA fragment from two contiguous 16-byte chunks.
// Per CDNA5 ISA 16-bit layout: lane<16 holds K=[koff..koff+8)+[koff+16..koff+24)
// with koff = (lane>=16)*8 handled by the caller via the pointer.
__device__ __forceinline__ v16h make_frag(const h16* p) {
  v8h lo = *(const v8h*)(p);
  v8h hi = *(const v8h*)(p + 16);
  v16h f;
#pragma unroll
  for (int i = 0; i < 8; ++i) { f[i] = lo[i]; f[i + 8] = hi[i]; }
  return f;
}

#if USE_TDM
typedef __attribute__((ext_vector_type(4))) unsigned int u32x4;
typedef __attribute__((ext_vector_type(8))) int          i32x8;
typedef __attribute__((ext_vector_type(4))) int          i32x4;

// Issue one TDM 2D tile load: tile (rows x 32 halves) from a [Mrows x K]
// f16 row-major tensor into LDS at byte offset lds_off, padding each 64B row
// with 16B (-> 40-half padded LDS stride). D# layout per CDNA5 ISA sec.8.
__device__ __forceinline__ void tdm_tile_load(const h16* gptr, unsigned lds_off,
                                              int rows, int K, int Mrows) {
  const unsigned long long ga = (unsigned long long)(uintptr_t)gptr;
  u32x4 g0;
  g0[0] = 1u;                                   // count=1 (valid), user mode
  g0[1] = lds_off;                              // lds_addr (bytes)
  g0[2] = (unsigned)ga;                         // global_addr[31:0]
  g0[3] = (unsigned)((ga >> 32) & 0x01FFFFFFull) | (2u << 30);  // addr[56:32]|type=2

  i32x8 g1;
  g1[0] = (int)((1u << 16)      // data_size = 2 bytes
              | (1u << 20)      // pad_enable
              | (3u << 22)      // pad_interval: 16 DWORDs (= 64B row)
              | (3u << 25));    // pad_amount:   4 DWORDs (= 16B pad)
  g1[1] = (int)(((unsigned)K & 0xffffu) << 16);                           // td0[15:0]
  g1[2] = (int)(((unsigned)K >> 16) | (((unsigned)Mrows & 0xffffu) << 16)); // td0[31:16]|td1[15:0]
  g1[3] = (int)(((unsigned)Mrows >> 16) | (32u << 16));                   // td1[31:16]|tile_dim0=32
  g1[4] = rows;     // tile_dim1 ; tile_dim2 = 0
  g1[5] = K;        // tensor_dim0_stride[31:0]
  g1[6] = 0;        // stride[47:32]=0, dim1_stride lo=0
  g1[7] = 0;
  i32x4 z4; z4[0] = z4[1] = z4[2] = z4[3] = 0;  // groups 2/3 unused (2D tensor)
  i32x8 z8;
#pragma unroll
  for (int i = 0; i < 8; ++i) z8[i] = 0;
  __builtin_amdgcn_tensor_load_to_lds(g0, g1, z4, z4, z8, 0);
}
#endif

// ---------------- pack kernels ----------------
__global__ void cvt_f32_to_f16(const float* __restrict__ src, h16* __restrict__ dst, int n) {
  int i = blockIdx.x * blockDim.x + threadIdx.x;
  if (i < n) dst[i] = (h16)src[i];
}

// src [R][Cc] fp32 row-major -> dst [Cc][R] f16 (i.e. transposed)
__global__ void transpose_to_f16(const float* __restrict__ src, h16* __restrict__ dst,
                                 int R, int Cc) {
  int i = blockIdx.x * blockDim.x + threadIdx.x;
  if (i < R * Cc) {
    int r = i / Cc, c = i - r * Cc;
    dst[(size_t)c * R + r] = (h16)src[i];
  }
}

// ---------------- tiled WMMA GEMM: O = A[M,K] * Bt[N,K]^T + bias ----------------
// outH != null : f16 output.  outF != null : fp32 output with +xres residual.
__device__ __forceinline__ void gemm_body(const h16* __restrict__ A,
                                          const h16* __restrict__ Bt,
                                          const float* __restrict__ bias,
                                          h16* __restrict__ outH,
                                          float* __restrict__ outF,
                                          const float* __restrict__ xres,
                                          int M, int Nout, int K) {
  constexpr int BM = 128, BN = 64, LSTR = 40;   // padded LDS stride (halves)
  constexpr int A_H = BM * LSTR;                // halves per A buffer
  constexpr int B_H = BN * LSTR;                // halves per B buffer

  const int tid  = threadIdx.x;
  const int lane = tid & 31;
  const int w    = tid >> 5;                 // 8 waves
  const int wm   = (w & 3) * 32;             // wave tile origin in M
  const int wn   = (w >> 2) * 32;            // wave tile origin in N
  const int m0   = blockIdx.x * BM;
  const int n0   = blockIdx.y * BN;
  const int lrow = lane & 15;
  const int koff = (lane >= 16) ? 8 : 0;
  const int rbase = (lane >= 16) ? 8 : 0;

  v8f acc[2][2];
#pragma unroll
  for (int i = 0; i < 2; ++i)
#pragma unroll
    for (int j = 0; j < 2; ++j)
#pragma unroll
      for (int e = 0; e < 8; ++e) acc[i][j][e] = 0.f;

  const int nk = K / 32;

#if USE_TDM
  // ---- TDM double-buffered pipeline: TENSOR_LOAD_TO_LDS with LDS row padding ----
  __shared__ h16 smem[2 * A_H + 2 * B_H];
  if (w == 0) {  // one wave owns the DMA descriptors (per-wave TENSORcnt)
    tdm_tile_load(A  + (size_t)m0 * K, 0u, BM, K, M);
    tdm_tile_load(Bt + (size_t)n0 * K, (unsigned)(2 * A_H * sizeof(h16)), BN, K, Nout);
  }
  for (int i = 0; i < nk; ++i) {
    const int cur = i & 1, nxt = cur ^ 1;
    __syncthreads();  // everyone done reading buffer `nxt` from iteration i-1
    if (w == 0) {
      if (i + 1 < nk) {
        const int k1 = (i + 1) * 32;
        tdm_tile_load(A  + (size_t)m0 * K + k1,
                      (unsigned)(nxt * A_H * sizeof(h16)), BM, K, M);
        tdm_tile_load(Bt + (size_t)n0 * K + k1,
                      (unsigned)((2 * A_H + nxt * B_H) * sizeof(h16)), BN, K, Nout);
        __builtin_amdgcn_s_wait_tensorcnt(2);  // current buffer's 2 DMAs landed
      } else {
        __builtin_amdgcn_s_wait_tensorcnt(0);
      }
    }
    __syncthreads();  // release: buffer `cur` visible to all waves

    const int aB = cur * A_H;
    const int bB = 2 * A_H + cur * B_H;
    v16h af[2], bf[2];
#pragma unroll
    for (int t = 0; t < 2; ++t)
      af[t] = make_frag(&smem[aB + (wm + t * 16 + lrow) * LSTR + koff]);
#pragma unroll
    for (int t = 0; t < 2; ++t)
      bf[t] = make_frag(&smem[bB + (wn + t * 16 + lrow) * LSTR + koff]);
#pragma unroll
    for (int i2 = 0; i2 < 2; ++i2)
#pragma unroll
      for (int j = 0; j < 2; ++j)
        acc[i2][j] = WMMA_F32_F16(af[i2], bf[j], acc[i2][j]);
  }
#else
  // ---- fallback: synchronous VGPR-staged tiles ----
  __shared__ h16 smem[A_H + B_H];
  for (int i = 0; i < nk; ++i) {
    const int k0 = i * 32;
    __syncthreads();
    {  // A tile: 128 rows x 32 halves
      int r = tid >> 1, part = (tid & 1) * 16;
      const v8h* sa = (const v8h*)(A + (size_t)(m0 + r) * K + k0 + part);
      v8h a0 = sa[0], a1 = sa[1];
      *(v8h*)&smem[r * LSTR + part]     = a0;
      *(v8h*)&smem[r * LSTR + part + 8] = a1;
    }
    if (tid < 128) {  // B tile: 64 rows x 32 halves
      int r = tid >> 1, part = (tid & 1) * 16;
      const v8h* sb = (const v8h*)(Bt + (size_t)(n0 + r) * K + k0 + part);
      v8h b0 = sb[0], b1 = sb[1];
      *(v8h*)&smem[A_H + r * LSTR + part]     = b0;
      *(v8h*)&smem[A_H + r * LSTR + part + 8] = b1;
    }
    __syncthreads();
    v16h af[2], bf[2];
#pragma unroll
    for (int t = 0; t < 2; ++t)
      af[t] = make_frag(&smem[(wm + t * 16 + lrow) * LSTR + koff]);
#pragma unroll
    for (int t = 0; t < 2; ++t)
      bf[t] = make_frag(&smem[A_H + (wn + t * 16 + lrow) * LSTR + koff]);
#pragma unroll
    for (int i2 = 0; i2 < 2; ++i2)
#pragma unroll
      for (int j = 0; j < 2; ++j)
        acc[i2][j] = WMMA_F32_F16(af[i2], bf[j], acc[i2][j]);
  }
#endif

  // epilogue (C layout: VGPR e -> row e+rbase, col = lane%16)
#pragma unroll
  for (int i = 0; i < 2; ++i) {
#pragma unroll
    for (int j = 0; j < 2; ++j) {
      const int col = n0 + wn + j * 16 + lrow;
      const float bv = bias[col];
#pragma unroll
      for (int e = 0; e < 8; ++e) {
        const int row = m0 + wm + i * 16 + rbase + e;
        const size_t idx = (size_t)row * Nout + col;
        const float v = acc[i][j][e] + bv;
        if (outF) outF[idx] = v + xres[idx];
        else      outH[idx] = (h16)v;
      }
    }
  }
}

__global__ __launch_bounds__(256)
void gemm_proj(const h16* __restrict__ A,
               const h16* __restrict__ Bt0, const h16* __restrict__ Bt1, const h16* __restrict__ Bt2,
               const float* __restrict__ b0, const float* __restrict__ b1, const float* __restrict__ b2,
               h16* __restrict__ O0, h16* __restrict__ O1, h16* __restrict__ O2,
               int M, int Nout, int K) {
  const h16*   Bt   = (blockIdx.z == 0) ? Bt0 : (blockIdx.z == 1) ? Bt1 : Bt2;
  const float* bias = (blockIdx.z == 0) ? b0  : (blockIdx.z == 1) ? b1  : b2;
  h16*         Out  = (blockIdx.z == 0) ? O0  : (blockIdx.z == 1) ? O1  : O2;
  gemm_body(A, Bt, bias, Out, nullptr, nullptr, M, Nout, K);
}

__global__ __launch_bounds__(256)
void gemm_out(const h16* __restrict__ A, const h16* __restrict__ Bt,
              const float* __restrict__ bias, const float* __restrict__ xres,
              float* __restrict__ outF, int M, int Nout, int K) {
  gemm_body(A, Bt, bias, nullptr, outF, xres, M, Nout, K);
}

// ---------------- fused flash-style attention ----------------
// grid = (N/128, B); 8 waves/block; wave owns 16 rows x full D output.
__global__ __launch_bounds__(256)
void attn_fused(const h16* __restrict__ Th, const h16* __restrict__ Ph,
                const h16* __restrict__ Gh, const float* __restrict__ adj,
                h16* __restrict__ Oh) {
  constexpr int GTS = 40;  // gT padded n-stride (halves)
  constexpr int PSS = 40;  // P scratch padded stride (halves)
  __shared__ h16 gT[kD * GTS];        // g tile transposed: [d][n0..n0+31]
  __shared__ h16 Ps[8 * 16 * PSS];    // per-wave P re-layout scratch

  const int tid   = threadIdx.x;
  const int lane  = tid & 31;
  const int w     = tid >> 5;
  const int b     = blockIdx.y;
  const int mrow0 = blockIdx.x * 128 + w * 16;
  const int lrow  = lane & 15;
  const int koff  = (lane >= 16) ? 8 : 0;
  const int rbase = (lane >= 16) ? 8 : 0;
  const size_t bN = (size_t)b * kN;

  v8f Oacc[24];
#pragma unroll
  for (int t = 0; t < 24; ++t)
#pragma unroll
    for (int e = 0; e < 8; ++e) Oacc[t][e] = 0.f;

  float mrun[8], lrun[8];
#pragma unroll
  for (int e = 0; e < 8; ++e) { mrun[e] = -1e30f; lrun[e] = 0.f; }

  const h16* Arow = Th + (bN + mrow0 + lrow) * kD;  // theta row for this lane
  h16* myPs = &Ps[w * 16 * PSS];

  for (int nt = 0; nt < kN / 32; ++nt) {
    const int n0 = nt * 32;
    __syncthreads();
    {  // load g rows [n0, n0+32) transposed into LDS: 8 threads per row
      const int n = tid >> 3;
      const int dbase = (tid & 7) * 48;
      const v8h* src = (const v8h*)(Gh + (bN + n0 + n) * kD + dbase);
#pragma unroll
      for (int c2 = 0; c2 < 6; ++c2) {
        v8h v = src[c2];
#pragma unroll
        for (int e = 0; e < 8; ++e) gT[(dbase + c2 * 8 + e) * GTS + n] = v[e];
      }
    }
    __syncthreads();

    // S(16x32) = theta(16xD) @ phi(n-tile)^T ; phi row-major == Bt layout
    v8f s0, s1;
#pragma unroll
    for (int e = 0; e < 8; ++e) { s0[e] = 0.f; s1[e] = 0.f; }
    const h16* Brow0 = Ph + (bN + n0 + lrow) * kD;
    const h16* Brow1 = Ph + (bN + n0 + 16 + lrow) * kD;
    if (nt + 1 < kN / 32) {  // prefetch next tile's phi stream + adj row
      __builtin_prefetch(Brow0 + 32 * kD, 0, 0);
      __builtin_prefetch(Brow1 + 32 * kD, 0, 0);
      __builtin_prefetch(&adj[((size_t)b * kN + mrow0 + rbase) * kN + n0 + 32], 0, 0);
    }
#pragma unroll
    for (int ks = 0; ks < kD / 32; ++ks) {
      const int k0 = ks * 32;
      v16h af  = make_frag(Arow  + k0 + koff);
      v16h bf0 = make_frag(Brow0 + k0 + koff);
      v16h bf1 = make_frag(Brow1 + k0 + koff);
      s0 = WMMA_F32_F16(af, bf0, s0);
      s1 = WMMA_F32_F16(af, bf1, s1);
    }

    // elementwise adjacency mask + online softmax
    float sv[2][8], tmax[8];
#pragma unroll
    for (int e = 0; e < 8; ++e) tmax[e] = -1e30f;
#pragma unroll
    for (int t = 0; t < 2; ++t) {
#pragma unroll
      for (int e = 0; e < 8; ++e) {
        const int row = mrow0 + rbase + e;
        const int col = n0 + t * 16 + lrow;
        const float raw = (t == 0) ? s0[e] : s1[e];
        const float v = raw * adj[((size_t)b * kN + row) * kN + col];
        sv[t][e] = v;
        tmax[e] = fmaxf(tmax[e], v);
      }
    }
#pragma unroll
    for (int e = 0; e < 8; ++e)
#pragma unroll
      for (int off = 1; off < 16; off <<= 1)
        tmax[e] = fmaxf(tmax[e], __shfl_xor(tmax[e], off, 16));

    float scale[8], rsum[8];
#pragma unroll
    for (int e = 0; e < 8; ++e) {
      const float nm = fmaxf(mrun[e], tmax[e]);
      scale[e] = __expf(mrun[e] - nm);
      mrun[e] = nm;
      rsum[e] = 0.f;
    }
    // P = exp(S - m) -> per-wave LDS scratch (converts C-layout -> A-layout)
#pragma unroll
    for (int t = 0; t < 2; ++t) {
#pragma unroll
      for (int e = 0; e < 8; ++e) {
        const float p = __expf(sv[t][e] - mrun[e]);
        rsum[e] += p;
        myPs[(rbase + e) * PSS + t * 16 + lrow] = (h16)p;
      }
    }
#pragma unroll
    for (int e = 0; e < 8; ++e) {
#pragma unroll
      for (int off = 1; off < 16; off <<= 1)
        rsum[e] += __shfl_xor(rsum[e], off, 16);
      lrun[e] = lrun[e] * scale[e] + rsum[e];
    }
    // rescale running output accumulator
#pragma unroll
    for (int t = 0; t < 24; ++t)
#pragma unroll
      for (int e = 0; e < 8; ++e) Oacc[t][e] *= scale[e];

    asm volatile("s_wait_dscnt 0" ::: "memory");  // wave-local LDS RAW fence

    // O += P(16x32) @ g(32xD)
    v16h pf = make_frag(&myPs[lrow * PSS + koff]);
#pragma unroll
    for (int dt = 0; dt < 24; ++dt) {
      v16h gf = make_frag(&gT[(dt * 16 + lrow) * GTS + koff]);
      Oacc[dt] = WMMA_F32_F16(pf, gf, Oacc[dt]);
    }
  }

  // finalize: divide by softmax denominator, emit f16 for the output GEMM
#pragma unroll
  for (int e = 0; e < 8; ++e) lrun[e] = 1.f / lrun[e];
#pragma unroll
  for (int dt = 0; dt < 24; ++dt) {
#pragma unroll
    for (int e = 0; e < 8; ++e) {
      const int row = mrow0 + rbase + e;
      const int col = dt * 16 + lrow;
      Oh[(bN + row) * kD + col] = (h16)(Oacc[dt][e] * lrun[e]);
    }
  }
}

// ---------------- host launch ----------------
extern "C" void kernel_launch(void* const* d_in, const int* in_sizes, int n_in,
                              void* d_out, int out_size, void* d_ws, size_t ws_size,
                              hipStream_t stream) {
  const float* x    = (const float*)d_in[0];
  const float* adj  = (const float*)d_in[1];
  const float* W_th = (const float*)d_in[2];
  const float* b_th = (const float*)d_in[3];
  const float* W_ph = (const float*)d_in[4];
  const float* b_ph = (const float*)d_in[5];
  const float* W_g  = (const float*)d_in[6];
  const float* b_g  = (const float*)d_in[7];
  const float* W_o  = (const float*)d_in[8];
  const float* b_o  = (const float*)d_in[9];
  float* out = (float*)d_out;

  char* ws = (char*)d_ws;
  size_t off = 0;
  auto alloc = [&](size_t bytes) -> char* {
    char* p = ws + off;
    off = (off + bytes + 255) & ~(size_t)255;
    return p;
  };
  h16* xh   = (h16*)alloc((size_t)kMN * kC * sizeof(h16));
  h16* wtT  = (h16*)alloc((size_t)kD * kC * sizeof(h16));
  h16* wtP  = (h16*)alloc((size_t)kD * kC * sizeof(h16));
  h16* wtG  = (h16*)alloc((size_t)kD * kC * sizeof(h16));
  h16* wtO  = (h16*)alloc((size_t)kC * kD * sizeof(h16));
  h16* thH  = (h16*)alloc((size_t)kMN * kD * sizeof(h16));
  h16* phH  = (h16*)alloc((size_t)kMN * kD * sizeof(h16));
  h16* gH   = (h16*)alloc((size_t)kMN * kD * sizeof(h16));
  h16* aoH  = (h16*)alloc((size_t)kMN * kD * sizeof(h16));

  // 1) pack x and weights to f16 (weights transposed to [Nout][K])
  {
    const int n = kMN * kC;
    cvt_f32_to_f16<<<(n + 255) / 256, 256, 0, stream>>>(x, xh, n);
    const int nw = kC * kD;
    transpose_to_f16<<<(nw + 255) / 256, 256, 0, stream>>>(W_th, wtT, kC, kD);
    transpose_to_f16<<<(nw + 255) / 256, 256, 0, stream>>>(W_ph, wtP, kC, kD);
    transpose_to_f16<<<(nw + 255) / 256, 256, 0, stream>>>(W_g,  wtG, kC, kD);
    transpose_to_f16<<<(nw + 255) / 256, 256, 0, stream>>>(W_o,  wtO, kD, kC);
  }

  // 2) three fused projections: theta/phi/g = x @ W + b  (f16 out)
  {
    dim3 grid(kMN / 128, kD / 64, 3);
    gemm_proj<<<grid, 256, 0, stream>>>(xh, wtT, wtP, wtG, b_th, b_ph, b_g,
                                        thH, phH, gH, kMN, kD, kC);
  }

  // 3) fused masked attention + softmax + P@g (flash-style, no score spill)
  {
    dim3 grid(kN / 128, kB, 1);
    attn_fused<<<grid, 256, 0, stream>>>(thH, phH, gH, adj, aoH);
  }

  // 4) output projection + bias + residual
  {
    dim3 grid(kMN / 128, kC / 64, 1);
    gemm_out<<<grid, 256, 0, stream>>>(aoH, wtO, b_o, x, out, kMN, kC, kD);
  }

  (void)in_sizes; (void)n_in; (void)out_size; (void)ws_size;
}
